// StructFormer_4698694222089
// MI455X (gfx1250) — compile-verified, matrix operand order checked by reading, not run on hardware
//
#include <hip/hip_runtime.h>
#include <hip/hip_bf16.h>
#include <cstdint>
#include <cstddef>

// ---- model constants (fixed by reference) ----
#define B_   8
#define L_   128
#define H_   512
#define V_   10000
#define NL_  8
#define NH_  8
#define HD_  64
#define FF_  2048
#define NT_  (B_*L_)          // 1024 tokens
#define NEGV (-1e9f)
#define POSVAL (1e9f)

typedef __attribute__((ext_vector_type(16))) _Float16 v16h;
typedef __attribute__((ext_vector_type(8)))  _Float16 v8h;
typedef __attribute__((ext_vector_type(8)))  float    v8f;

// ---------------------------------------------------------------------------
// WMMA fragment helpers (CDNA5 16-bit 16x16x32 layout, ISA 7.12.2):
// lane = hh*16+p : holds K = [hh*8 .. hh*8+7] and [16+hh*8 .. 16+hh*8+7]
// of row (A) / column-of-Bt (B) index p.
// ---------------------------------------------------------------------------
static __device__ __forceinline__ v16h load_frag(const _Float16* __restrict__ rowk, int hh) {
  v8h lo = *(const v8h*)(rowk + hh*8);
  v8h hi = *(const v8h*)(rowk + 16 + hh*8);
  v16h r;
#pragma unroll
  for (int i = 0; i < 8; ++i) { r[i] = lo[i]; r[i+8] = hi[i]; }
  return r;
}
#define WMMA16(acc,a,b) \
  (acc) = __builtin_amdgcn_wmma_f32_16x16x32_f16(false,(a),false,(b),(short)0,(acc),false,false)

// ---------------------------------------------------------------------------
// Generic GEMM, register-blocked 16x64 per wave (4 accumulators, A-frag reuse):
//   C[M][N](f32,ldc) = A[M][K](f16,lda) * Bt[N][K]^T(f16,ldb) + bias
// epi: 0=store  1=accumulate (residual)  2=leaky_relu(0.01)
// K multiple of 32; M,N multiples of 16 (ragged N-edge handled by clamp+guard,
// wave-uniform so EXEC stays all-ones for WMMA).
// ---------------------------------------------------------------------------
__global__ void k_gemm(const _Float16* __restrict__ A, int lda,
                       const _Float16* __restrict__ Bt, int ldb,
                       const float* __restrict__ bias,
                       float* __restrict__ C, int ldc,
                       int M, int Nc, int K, int epi) {
  int wv   = (int)((blockIdx.x * blockDim.x + threadIdx.x) >> 5);
  int lane = threadIdx.x & 31;
  int Ntl = Nc >> 4;                  // 16-col tiles
  int Ng  = (Ntl + 3) >> 2;           // groups of 4 tiles (64 cols)
  int tiles = (M >> 4) * Ng;
  if (wv >= tiles) return;
  int row0 = (wv / Ng) << 4;
  int ct0  = (wv % Ng) << 2;          // first 16-col tile of this group
  int c0 = (ct0     < Ntl) ? ct0     : Ntl - 1;
  int c1 = (ct0 + 1 < Ntl) ? ct0 + 1 : Ntl - 1;
  int c2 = (ct0 + 2 < Ntl) ? ct0 + 2 : Ntl - 1;
  int c3 = (ct0 + 3 < Ntl) ? ct0 + 3 : Ntl - 1;
  int p = lane & 15, hh = lane >> 4;
  const _Float16* ap  = A  + (size_t)(row0 + p) * lda;
  const _Float16* bp0 = Bt + (size_t)(c0 * 16 + p) * ldb;
  const _Float16* bp1 = Bt + (size_t)(c1 * 16 + p) * ldb;
  const _Float16* bp2 = Bt + (size_t)(c2 * 16 + p) * ldb;
  const _Float16* bp3 = Bt + (size_t)(c3 * 16 + p) * ldb;
  v8f a0 = {}, a1 = {}, a2 = {}, a3 = {};
  for (int kk = 0; kk < K; kk += 32) {
    __builtin_prefetch(ap + kk + 256, 0, 0);   // gfx1250 global_prefetch_b8
    v16h af  = load_frag(ap  + kk, hh);
    v16h b0f = load_frag(bp0 + kk, hh);
    v16h b1f = load_frag(bp1 + kk, hh);
    v16h b2f = load_frag(bp2 + kk, hh);
    v16h b3f = load_frag(bp3 + kk, hh);
    WMMA16(a0, af, b0f);
    WMMA16(a1, af, b1f);
    WMMA16(a2, af, b2f);
    WMMA16(a3, af, b3f);
  }
  int n = lane & 15, mb = hh * 8;
#pragma unroll
  for (int j = 0; j < 4; ++j) {
    if (ct0 + j >= Ntl) break;                 // ragged edge (uniform)
    int col0 = (ct0 + j) * 16;
    float bv = bias ? bias[col0 + n] : 0.f;
    v8f acc = (j == 0) ? a0 : (j == 1) ? a1 : (j == 2) ? a2 : a3;
#pragma unroll
    for (int v = 0; v < 8; ++v) {
      float val = acc[v] + bv;
      size_t idx = (size_t)(row0 + mb + v) * ldc + col0 + n;
      if (epi == 1)      C[idx] += val;
      else if (epi == 2) C[idx]  = (val > 0.f) ? val : 0.01f * val;
      else               C[idx]  = val;
    }
  }
}

// ---------------------------------------------------------------------------
// Conv1d as GEMM over taps, 16x64 per wave:
// out[(b,l)][oc] = bias + sum_k hp[b][l+k] . Wt[k][oc][:]
// hp: [B][Lp][H] f16 (zero-padded), Wt: [KW][H(oc)][H(c)] f16
// ---------------------------------------------------------------------------
__global__ void k_conv_gemm(const _Float16* __restrict__ hp, int Lp,
                            const _Float16* __restrict__ Wt,
                            const float* __restrict__ bias,
                            float* __restrict__ out, int KW) {
  int wv   = (int)((blockIdx.x * blockDim.x + threadIdx.x) >> 5);
  int lane = threadIdx.x & 31;
  const int Ng = H_ / 64;                         // 8 groups of 64 cols
  const int tiles = (NT_ / 16) * Ng;              // 512
  if (wv >= tiles) return;
  int row0 = (wv / Ng) << 4;
  int col0 = (wv % Ng) << 6;                      // 64-col group base
  int b = row0 >> 7, l0 = row0 & (L_ - 1);
  int p = lane & 15, hh = lane >> 4;
  v8f a0 = {}, a1 = {}, a2 = {}, a3 = {};
  for (int k = 0; k < KW; ++k) {
    const _Float16* ap  = hp + (size_t)(b * Lp + l0 + p + k) * H_;
    const _Float16* wb  = Wt + (size_t)k * H_ * H_;
    const _Float16* bp0 = wb + (size_t)(col0 +  0 + p) * H_;
    const _Float16* bp1 = wb + (size_t)(col0 + 16 + p) * H_;
    const _Float16* bp2 = wb + (size_t)(col0 + 32 + p) * H_;
    const _Float16* bp3 = wb + (size_t)(col0 + 48 + p) * H_;
    for (int kk = 0; kk < H_; kk += 32) {
      __builtin_prefetch(ap + kk + 256, 0, 0);
      v16h af  = load_frag(ap  + kk, hh);
      v16h b0f = load_frag(bp0 + kk, hh);
      v16h b1f = load_frag(bp1 + kk, hh);
      v16h b2f = load_frag(bp2 + kk, hh);
      v16h b3f = load_frag(bp3 + kk, hh);
      WMMA16(a0, af, b0f);
      WMMA16(a1, af, b1f);
      WMMA16(a2, af, b2f);
      WMMA16(a3, af, b3f);
    }
  }
  int n = lane & 15, mb = hh * 8;
#pragma unroll
  for (int j = 0; j < 4; ++j) {
    int col = col0 + j * 16 + n;
    float bv = bias[col];
    v8f acc = (j == 0) ? a0 : (j == 1) ? a1 : (j == 2) ? a2 : a3;
#pragma unroll
    for (int v = 0; v < 8; ++v)
      out[(size_t)(row0 + mb + v) * H_ + col] = acc[v] + bv;
  }
}

// ---------------------------------------------------------------------------
// Attention scores, 16(l) x 64(m) per wave:
// w16[b,nh,l,m] = sigmoid(q.k + kpm[b,m]) * (rw0*head[b,l,m] + rw1*head[b,m,l])
// qh,kh: [B*NH][L][HD] f16 (q pre-scaled by hd^-0.5)
// ---------------------------------------------------------------------------
__global__ void k_attn_score(const _Float16* __restrict__ qh, const _Float16* __restrict__ kh,
                             const float* __restrict__ head, const float* __restrict__ maskNeg,
                             const float* __restrict__ relw, int layer,
                             _Float16* __restrict__ w16) {
  int wv   = (int)((blockIdx.x * blockDim.x + threadIdx.x) >> 5);
  int lane = threadIdx.x & 31;
  const int TP = (L_/16) * (L_/64);               // 8 x 2 = 16 per (b,nh)
  if (wv >= B_*NH_*TP) return;
  int bh = wv / TP, tile = wv % TP;
  int b = bh >> 3, nh = bh & 7;
  int row0 = (tile / (L_/64)) << 4;               // l tile
  int col0 = (tile % (L_/64)) << 6;               // m group (64)
  int p = lane & 15, hh = lane >> 4;
  const _Float16* ap  = qh + (size_t)(bh * L_ + row0 + p) * HD_;
  const _Float16* bp0 = kh + (size_t)(bh * L_ + col0 +  0 + p) * HD_;
  const _Float16* bp1 = kh + (size_t)(bh * L_ + col0 + 16 + p) * HD_;
  const _Float16* bp2 = kh + (size_t)(bh * L_ + col0 + 32 + p) * HD_;
  const _Float16* bp3 = kh + (size_t)(bh * L_ + col0 + 48 + p) * HD_;
  v8f a0 = {}, a1 = {}, a2 = {}, a3 = {};
  for (int kk = 0; kk < HD_; kk += 32) {
    v16h af  = load_frag(ap  + kk, hh);
    v16h b0f = load_frag(bp0 + kk, hh);
    v16h b1f = load_frag(bp1 + kk, hh);
    v16h b2f = load_frag(bp2 + kk, hh);
    v16h b3f = load_frag(bp3 + kk, hh);
    WMMA16(a0, af, b0f);
    WMMA16(a1, af, b1f);
    WMMA16(a2, af, b2f);
    WMMA16(a3, af, b3f);
  }
  float rw0 = relw[(layer*NH_ + nh)*2 + 0];
  float rw1 = relw[(layer*NH_ + nh)*2 + 1];
  int n = lane & 15, mb = hh * 8;
#pragma unroll
  for (int j = 0; j < 4; ++j) {
    int mcol = col0 + j * 16 + n;
    float kpm = maskNeg[b * L_ + mcol];
    v8f acc = (j == 0) ? a0 : (j == 1) ? a1 : (j == 2) ? a2 : a3;
#pragma unroll
    for (int v = 0; v < 8; ++v) {
      int lrow = row0 + mb + v;
      float s = 1.f / (1.f + expf(-(acc[v] + kpm)));
      float dep = rw0 * head[((size_t)b * L_ + lrow) * L_ + mcol]
                + rw1 * head[((size_t)b * L_ + mcol) * L_ + lrow];
      w16[((size_t)bh * L_ + lrow) * L_ + mcol] = (_Float16)(s * dep);
    }
  }
}

// o32[(b*L+l)][nh*HD+d] = sum_m w16[bh,l,m] * vT[bh,d,m] ; 16(l) x 64(d) per wave
__global__ void k_attn_out(const _Float16* __restrict__ w16, const _Float16* __restrict__ vT,
                           float* __restrict__ o32) {
  int wv   = (int)((blockIdx.x * blockDim.x + threadIdx.x) >> 5);
  int lane = threadIdx.x & 31;
  const int TP = (L_/16);                         // 8 per (b,nh), full HD=64 width
  if (wv >= B_*NH_*TP) return;
  int bh = wv / TP, tile = wv % TP;
  int b = bh >> 3, nh = bh & 7;
  int row0 = tile << 4;                           // l
  int p = lane & 15, hh = lane >> 4;
  const _Float16* ap  = w16 + (size_t)(bh * L_ + row0 + p) * L_;
  const _Float16* bp0 = vT  + (size_t)(bh * HD_ +  0 + p) * L_;
  const _Float16* bp1 = vT  + (size_t)(bh * HD_ + 16 + p) * L_;
  const _Float16* bp2 = vT  + (size_t)(bh * HD_ + 32 + p) * L_;
  const _Float16* bp3 = vT  + (size_t)(bh * HD_ + 48 + p) * L_;
  v8f a0 = {}, a1 = {}, a2 = {}, a3 = {};
  for (int kk = 0; kk < L_; kk += 32) {
    v16h af  = load_frag(ap  + kk, hh);
    v16h b0f = load_frag(bp0 + kk, hh);
    v16h b1f = load_frag(bp1 + kk, hh);
    v16h b2f = load_frag(bp2 + kk, hh);
    v16h b3f = load_frag(bp3 + kk, hh);
    WMMA16(a0, af, b0f);
    WMMA16(a1, af, b1f);
    WMMA16(a2, af, b2f);
    WMMA16(a3, af, b3f);
  }
  int n = lane & 15, mb = hh * 8;
#pragma unroll
  for (int j = 0; j < 4; ++j) {
    v8f acc = (j == 0) ? a0 : (j == 1) ? a1 : (j == 2) ? a2 : a3;
#pragma unroll
    for (int v = 0; v < 8; ++v)
      o32[(size_t)(b * L_ + row0 + mb + v) * H_ + nh * HD_ + j * 16 + n] = acc[v];
  }
}

// ---------------------------------------------------------------------------
// elementwise / scan kernels
// ---------------------------------------------------------------------------
__global__ void k_embed(const int* __restrict__ x, const float* __restrict__ emb,
                        float* __restrict__ h, float* __restrict__ maskNeg) {
  int i = blockIdx.x * 256 + threadIdx.x;
  if (i >= NT_ * H_) return;
  int n = i >> 9, c = i & 511;
  h[i] = emb[(size_t)x[n] * H_ + c];
  if (c == 0) maskNeg[n] = (x[n] != 0) ? 0.f : NEGV;
}

__global__ void k_cvt(const float* __restrict__ a, _Float16* __restrict__ b, int n) {
  int i = blockIdx.x * 256 + threadIdx.x;
  if (i < n) b[i] = (_Float16)a[i];
}

// W[K][N] f32 -> Wt[N][K] f16, batched over blockIdx.z (stride K*N)
__global__ void k_tcvt(const float* __restrict__ W, _Float16* __restrict__ Wt, int K, int Nc) {
  __shared__ float tile[32][33];
  size_t zoff = (size_t)blockIdx.z * K * Nc;
  int k0 = blockIdx.x * 32, n0 = blockIdx.y * 32;
  int tx = threadIdx.x & 31, ty = threadIdx.x >> 5;   // 256 threads -> ty in 0..7
#pragma unroll
  for (int r = 0; r < 32; r += 8)
    tile[ty + r][tx] = W[zoff + (size_t)(k0 + ty + r) * Nc + n0 + tx];
  __syncthreads();
#pragma unroll
  for (int r = 0; r < 32; r += 8)
    Wt[zoff + (size_t)(n0 + ty + r) * K + k0 + tx] = (_Float16)tile[tx][ty + r];
}

__global__ void k_pack_pad(const float* __restrict__ h, const int* __restrict__ x,
                           _Float16* __restrict__ hp, int Lp, int PL, int applyMask) {
  int i = blockIdx.x * 256 + threadIdx.x;
  if (i >= B_ * Lp * H_) return;
  int c = i % H_;
  int rest = i / H_;
  int lp = rest % Lp, b = rest / Lp;
  int l = lp - PL;
  float v = 0.f;
  if (l >= 0 && l < L_) {
    v = h[(size_t)(b * L_ + l) * H_ + c];
    if (applyMask && x[b * L_ + l] == 0) v = 0.f;
  }
  hp[i] = (_Float16)v;
}

// rowwise LayerNorm over H, optional affine + tanh
__global__ void k_ln(const float* __restrict__ in, float* __restrict__ out,
                     const float* __restrict__ g, const float* __restrict__ beta, int doTanh) {
  int row = blockIdx.x;
  const float* r = in + (size_t)row * H_;
  float s = 0.f, s2 = 0.f;
  for (int c = threadIdx.x; c < H_; c += 256) { float v = r[c]; s += v; s2 += v * v; }
  __shared__ float sa[256], sb[256];
  sa[threadIdx.x] = s; sb[threadIdx.x] = s2; __syncthreads();
  for (int off = 128; off; off >>= 1) {
    if (threadIdx.x < off) { sa[threadIdx.x] += sa[threadIdx.x+off]; sb[threadIdx.x] += sb[threadIdx.x+off]; }
    __syncthreads();
  }
  float mean = sa[0] / H_;
  float var  = sb[0] / H_ - mean * mean;
  float inv  = rsqrtf(fmaxf(var, 0.f) + 1e-5f);
  for (int c = threadIdx.x; c < H_; c += 256) {
    float v = (r[c] - mean) * inv;
    if (g) v = v * g[c] + beta[c];
    if (doTanh) v = tanhf(v);
    out[(size_t)row * H_ + c] = v;
  }
}

// out[row] = t[row,:].w + b, then mask transform (mode0: height, mode1: distance)
__global__ void k_rowdot(const float* __restrict__ t, const float* __restrict__ w,
                         const float* __restrict__ bptr, const int* __restrict__ x,
                         float* __restrict__ out, int mode) {
  int row = blockIdx.x;
  float s = 0.f;
  for (int c = threadIdx.x; c < H_; c += 256) s += t[(size_t)row * H_ + c] * w[c];
  __shared__ float sa[256];
  sa[threadIdx.x] = s; __syncthreads();
  for (int off = 128; off; off >>= 1) {
    if (threadIdx.x < off) sa[threadIdx.x] += sa[threadIdx.x + off];
    __syncthreads();
  }
  if (threadIdx.x == 0) {
    float val = sa[0] + bptr[0];
    if (mode == 0) out[row] = (x[row] != 0) ? val : NEGV;
    else {
      int l = row & (L_ - 1);
      bool ms = (l < L_ - 1) && (x[row + 1] != 0);
      out[row] = ms ? val : POSVAL;
    }
  }
}

__global__ void k_zero(float* p) { if (threadIdx.x == 0 && blockIdx.x == 0) p[0] = 0.f; }

// margin max: one block per (b,i); hm = running max of height over [i..j]
__global__ void k_margin(const float* __restrict__ height, const float* __restrict__ distance,
                         const int* __restrict__ x, float* __restrict__ gmax) {
  int bi = blockIdx.x;
  int b = bi >> 7, i = bi & (L_ - 1);
  int j = threadIdx.x;
  __shared__ float sh[L_];
  sh[j] = (j >= i) ? height[b * L_ + j] : -3e38f;
  __syncthreads();
  for (int off = 1; off < L_; off <<= 1) {
    float v = sh[j];
    float u = (j >= off) ? sh[j - off] : -3e38f;
    __syncthreads();
    sh[j] = fmaxf(v, u);
    __syncthreads();
  }
  float val = 0.f;
  if (j >= i) {
    float hm = sh[j];
    float dp = (i == 0) ? POSVAL : distance[b * L_ + i - 1];
    float dj = distance[b * L_ + j];
    float mg = fminf(fmaxf(dp - hm, 0.f), fmaxf(dj - hm, 0.f));
    bool mm = (i == 0) ? ((j < L_ - 1) && (x[b * L_ + j + 1] != 0))
                       : (x[b * L_ + j] != 0);
    val = mm ? mg : 0.f;
  }
  __syncthreads();
  sh[j] = val; __syncthreads();
  for (int off = 64; off; off >>= 1) {
    if (j < off) sh[j] = fmaxf(sh[j], sh[j + off]);
    __syncthreads();
  }
  if (j == 0) atomicMax(reinterpret_cast<unsigned int*>(gmax), __float_as_uint(sh[0]));
}

__global__ void k_dsub(float* __restrict__ d, const float* __restrict__ gmax) {
  int i = blockIdx.x * 256 + threadIdx.x;
  if (i < NT_) d[i] -= gmax[0];
}

// per-batch: e = exp((height - max)*s1), ce = inclusive prefix (ce[0]=0)
__global__ void k_esum(const float* __restrict__ height, const float* __restrict__ scaler,
                       float* __restrict__ e, float* __restrict__ ce) {
  int b = blockIdx.x, t = threadIdx.x;
  float s1 = expf(scaler[1]);
  __shared__ float hv[L_], mx[L_], ps[L_];
  hv[t] = height[b * L_ + t];
  mx[t] = hv[t]; __syncthreads();
  for (int off = 64; off; off >>= 1) {
    if (t < off) mx[t] = fmaxf(mx[t], mx[t + off]);
    __syncthreads();
  }
  float ev = expf((hv[t] - mx[0]) * s1);
  e[b * L_ + t] = ev;
  ps[t] = ev; __syncthreads();
  for (int off = 1; off < L_; off <<= 1) {
    float v = ps[t];
    float u = (t >= off) ? ps[t - off] : 0.f;
    __syncthreads();
    ps[t] = v + u;
    __syncthreads();
  }
  if (t == 0) ce[b * (L_ + 1)] = 0.f;
  ce[b * (L_ + 1) + t + 1] = ps[t];
}

// per (b,l): gamma row -> bml (suffix-min diff, j<=l) and bmr (prefix-min diff, j>=l)
__global__ void k_bmlbmr(const float* __restrict__ height, const float* __restrict__ distance,
                         const float* __restrict__ scaler,
                         float* __restrict__ bml, float* __restrict__ bmr) {
  int bl = blockIdx.x;
  int b = bl >> 7, l = bl & (L_ - 1);
  int j = threadIdx.x;
  float s0 = expf(scaler[0]);
  float hl = height[b * L_ + l];
  float g = 1.f / (1.f + expf((distance[b * L_ + j] - hl) * s0)); // sigmoid(-(d-h)*s0)
  __shared__ float sh[L_];
  // bml: suffix min of xl = (j<l ? g : 1)
  sh[j] = (j < l) ? g : 1.f; __syncthreads();
  for (int off = 1; off < L_; off <<= 1) {
    float v = sh[j];
    float u = (j + off < L_) ? sh[j + off] : 3e38f;
    __syncthreads();
    sh[j] = fminf(v, u);
    __syncthreads();
  }
  float cm = sh[j];
  float cmprev = (j > 0) ? sh[j - 1] : 0.f;
  __syncthreads();
  bml[(size_t)bl * L_ + j] = (j <= l) ? (cm - cmprev) : 0.f;
  // bmr: inclusive prefix min of xr = (j>=l ? g : 1); pm[j] = incl[j-1] (pm[0]=1)
  sh[j] = (j >= l) ? g : 1.f; __syncthreads();
  for (int off = 1; off < L_; off <<= 1) {
    float v = sh[j];
    float u = (j >= off) ? sh[j - off] : 3e38f;
    __syncthreads();
    sh[j] = fminf(v, u);
    __syncthreads();
  }
  float pmj  = (j > 0) ? sh[j - 1] : 1.f;
  float pmj1 = (j < L_ - 1) ? sh[j] : 0.f;
  bmr[(size_t)bl * L_ + j] = (j >= l) ? (pmj - pmj1) : 0.f;
}

// head[b,l,h] = e_h * [ h<l: sum_{i<=h} A_i | h>l: sum_{j>=h} B_j | h==l: 0 ]
// A_i = bml[i] * sum_{j>=l} bmr[j]/S(i,j),  B_j = bmr[j] * sum_{i<=l} bml[i]/S(i,j)
// S(i,j) = ce[j+1]-ce[i]   (softmax denominator over span [i..j])
__global__ void k_head(const float* __restrict__ bml, const float* __restrict__ bmr,
                       const float* __restrict__ e, const float* __restrict__ ce,
                       float* __restrict__ head) {
  int bl = blockIdx.x;
  int b = bl >> 7, l = bl & (L_ - 1);
  int t = threadIdx.x;
  __shared__ float sml[L_], smr[L_], sce[L_ + 1], sa[L_];
  sml[t] = bml[(size_t)bl * L_ + t];
  smr[t] = bmr[(size_t)bl * L_ + t];
  sce[t] = ce[b * (L_ + 1) + t];
  if (t == 0) sce[L_] = ce[b * (L_ + 1) + L_];
  __syncthreads();
  float A = 0.f, Bv = 0.f;
  if (t <= l) {
    float cei = sce[t], acc = 0.f;
    for (int j = l; j < L_; ++j) acc += smr[j] / fmaxf(sce[j + 1] - cei, 1e-30f);
    A = sml[t] * acc;
  }
  if (t >= l) {
    float cej = sce[t + 1], acc = 0.f;
    for (int i = 0; i <= l; ++i) acc += sml[i] / fmaxf(cej - sce[i], 1e-30f);
    Bv = smr[t] * acc;
  }
  // inclusive prefix sum of A
  sa[t] = A; __syncthreads();
  for (int off = 1; off < L_; off <<= 1) {
    float v = sa[t];
    float u = (t >= off) ? sa[t - off] : 0.f;
    __syncthreads();
    sa[t] = v + u;
    __syncthreads();
  }
  float PA = sa[t];
  __syncthreads();
  // inclusive suffix sum of B
  sa[t] = Bv; __syncthreads();
  for (int off = 1; off < L_; off <<= 1) {
    float v = sa[t];
    float u = (t + off < L_) ? sa[t + off] : 0.f;
    __syncthreads();
    sa[t] = v + u;
    __syncthreads();
  }
  float SB = sa[t];
  float ev = e[b * L_ + t];
  head[(size_t)bl * L_ + t] = (t < l) ? ev * PA : ((t > l) ? ev * SB : 0.f);
}

__global__ void k_relw(const float* __restrict__ raw, float* __restrict__ rw) {
  int i = threadIdx.x;
  if (i >= NL_ * NH_) return;
  float a = raw[2*i], b = raw[2*i+1];
  float m = fmaxf(a, b);
  float ea = expf(a - m), eb = expf(b - m);
  float s = ea + eb;
  rw[2*i] = ea / s; rw[2*i+1] = eb / s;
}

__global__ void k_pack_qkv(const float* __restrict__ q32, const float* __restrict__ k32,
                           const float* __restrict__ v32,
                           _Float16* __restrict__ qh, _Float16* __restrict__ kh,
                           _Float16* __restrict__ vT) {
  int i = blockIdx.x * 256 + threadIdx.x;
  if (i >= NT_ * H_) return;
  int n = i >> 9, c = i & 511;
  int b = n >> 7, l = n & (L_ - 1);
  int nh = c >> 6, d = c & 63;
  int bh = b * NH_ + nh;
  qh[((size_t)bh * L_ + l) * HD_ + d] = (_Float16)(q32[i] * 0.125f); // hd^-0.5
  kh[((size_t)bh * L_ + l) * HD_ + d] = (_Float16)k32[i];
  vT[((size_t)bh * HD_ + d) * L_ + l] = (_Float16)v32[i];
}

// ---------------------------------------------------------------------------
extern "C" void kernel_launch(void* const* d_in, const int* in_sizes, int n_in,
                              void* d_out, int out_size, void* d_ws, size_t ws_size,
                              hipStream_t stream) {
  (void)in_sizes; (void)n_in; (void)out_size; (void)ws_size;
  const int*   x       = (const int*)  d_in[0];
  const float* emb     = (const float*)d_in[2];
  const float* pconv_w = (const float*)d_in[3];
  const float* pconv_b = (const float*)d_in[4];
  const float* dconv_w = (const float*)d_in[5];
  const float* dconv_b = (const float*)d_in[6];
  const float* dlin_w  = (const float*)d_in[7];
  const float* dlin_b  = (const float*)d_in[8];
  const float* hw1     = (const float*)d_in[9];
  const float* hb1     = (const float*)d_in[10];
  const float* hw2     = (const float*)d_in[11];
  const float* hb2     = (const float*)d_in[12];
  const float* relraw  = (const float*)d_in[13];
  const float* scaler  = (const float*)d_in[14];
  const float* aln_g   = (const float*)d_in[15];
  const float* aln_b   = (const float*)d_in[16];
  const float* qw      = (const float*)d_in[17];
  const float* qb      = (const float*)d_in[18];
  const float* kw      = (const float*)d_in[19];
  const float* kb      = (const float*)d_in[20];
  const float* vw      = (const float*)d_in[21];
  const float* vb      = (const float*)d_in[22];
  const float* ow      = (const float*)d_in[23];
  const float* ob      = (const float*)d_in[24];
  const float* fln_g   = (const float*)d_in[25];
  const float* fln_b   = (const float*)d_in[26];
  const float* fw1     = (const float*)d_in[27];
  const float* fb1     = (const float*)d_in[28];
  const float* fw2     = (const float*)d_in[29];
  const float* fb2     = (const float*)d_in[30];
  const float* fing    = (const float*)d_in[31];
  const float* finb    = (const float*)d_in[32];
  const float* obias   = (const float*)d_in[33];
  float* out = (float*)d_out;

  // ---- workspace carve-out ----
  char* wp = (char*)d_ws;
  auto alloc = [&](size_t bytes) -> char* {
    char* r = wp; wp += (bytes + 255) & ~(size_t)255; return r;
  };
  float* hP     = (float*)alloc((size_t)NT_*H_*4);
  float* hT     = (float*)alloc((size_t)NT_*H_*4);
  float* z32    = (float*)alloc((size_t)NT_*H_*4);
  float* q32    = (float*)alloc((size_t)NT_*H_*4);
  float* k32    = (float*)alloc((size_t)NT_*H_*4);
  float* v32    = (float*)alloc((size_t)NT_*H_*4);
  float* o32    = (float*)alloc((size_t)NT_*H_*4);
  float* u32    = (float*)alloc((size_t)NT_*FF_*4);
  float* t32    = (float*)alloc((size_t)NT_*H_*4);
  float* height = (float*)alloc(NT_*4);
  float* dist   = (float*)alloc(NT_*4);
  float* maskNeg= (float*)alloc(NT_*4);
  float* evec   = (float*)alloc(NT_*4);
  float* ce     = (float*)alloc((size_t)B_*(L_+1)*4);
  float* relw   = (float*)alloc(NL_*NH_*2*4);
  float* gmax   = (float*)alloc(256);
  float* bml    = (float*)alloc((size_t)NT_*L_*4);
  float* bmr    = (float*)alloc((size_t)NT_*L_*4);
  float* head   = (float*)alloc((size_t)NT_*L_*4);
  _Float16* hp16  = (_Float16*)alloc((size_t)B_*(L_+8)*H_*2);
  _Float16* hp2   = (_Float16*)alloc((size_t)B_*(L_+1)*H_*2);
  _Float16* buf16 = (_Float16*)alloc((size_t)NT_*FF_*2);
  _Float16* qh    = (_Float16*)alloc((size_t)NT_*H_*2);
  _Float16* kh    = (_Float16*)alloc((size_t)NT_*H_*2);
  _Float16* vT    = (_Float16*)alloc((size_t)NT_*H_*2);
  _Float16* w16   = (_Float16*)alloc((size_t)B_*NH_*L_*L_*2);
  _Float16* WtConv= (_Float16*)alloc((size_t)9*H_*H_*2);
  _Float16* WtQ   = (_Float16*)alloc((size_t)H_*H_*2);
  _Float16* WtK   = (_Float16*)alloc((size_t)H_*H_*2);
  _Float16* WtV   = (_Float16*)alloc((size_t)H_*H_*2);
  _Float16* WtO   = (_Float16*)alloc((size_t)H_*H_*2);
  _Float16* WtF1  = (_Float16*)alloc((size_t)H_*FF_*2);
  _Float16* WtF2  = (_Float16*)alloc((size_t)FF_*H_*2);
  _Float16* WtH1  = (_Float16*)alloc((size_t)H_*H_*2);
  _Float16* WtD   = (_Float16*)alloc((size_t)2*H_*H_*2);
  _Float16* emb16 = (_Float16*)alloc((size_t)V_*H_*2);

  auto GEMM = [&](const _Float16* A, int lda, const _Float16* Bt, int ldb,
                  const float* bias, float* C, int ldc, int M, int Nc, int K, int epi) {
    int Ng = ((Nc/16) + 3) / 4;
    int tiles = (M/16) * Ng;
    int blocks = (tiles*32 + 255)/256;
    k_gemm<<<blocks, 256, 0, stream>>>(A, lda, Bt, ldb, bias, C, ldc, M, Nc, K, epi);
  };
  auto TCVT = [&](const float* W, _Float16* Wt, int K, int Nc, int Z) {
    k_tcvt<<<dim3(K/32, Nc/32, Z), 256, 0, stream>>>(W, Wt, K, Nc);
  };
  auto CVT = [&](const float* a, _Float16* b, int n) {
    k_cvt<<<(n + 255)/256, 256, 0, stream>>>(a, b, n);
  };

  const int NH32 = (NT_*H_ + 255)/256;

  // ---- prelude ----
  k_relw<<<1, 64, 0, stream>>>(relraw, relw);
  CVT(emb, emb16, V_*H_);
  k_embed<<<NH32, 256, 0, stream>>>(x, emb, hP, maskNeg);

  // ---- parser conv stack (4 layers, K=9) ----
  for (int i = 0; i < 4; ++i) {
    k_pack_pad<<<(B_*(L_+8)*H_ + 255)/256, 256, 0, stream>>>(hP, x, hp16, L_+8, 4, 1);
    TCVT(pconv_w + (size_t)i*9*H_*H_, WtConv, H_, H_, 9);
    {
      int tiles = (NT_/16)*(H_/64);
      k_conv_gemm<<<(tiles*32 + 255)/256, 256, 0, stream>>>(hp16, L_+8, WtConv,
                                                            pconv_b + i*H_, t32, 9);
    }
    k_ln<<<NT_, 256, 0, stream>>>(t32, hP, nullptr, nullptr, 1);
  }

  // ---- height head ----
  CVT(hP, buf16, NT_*H_);
  TCVT(hw1, WtH1, H_, H_, 1);
  GEMM(buf16, H_, WtH1, H_, hb1, t32, H_, NT_, H_, H_, 0);
  k_ln<<<NT_, 256, 0, stream>>>(t32, t32, nullptr, nullptr, 1);
  k_rowdot<<<NT_, 256, 0, stream>>>(t32, hw2, hb2, x, height, 0);

  // ---- distance head (K=2 shifted conv) ----
  k_pack_pad<<<(B_*(L_+1)*H_ + 255)/256, 256, 0, stream>>>(hP, x, hp2, L_+1, 0, 0);
  TCVT(dconv_w, WtD, H_, H_, 2);
  {
    int tiles = (NT_/16)*(H_/64);
    k_conv_gemm<<<(tiles*32 + 255)/256, 256, 0, stream>>>(hp2, L_+1, WtD, dconv_b, t32, 2);
  }
  k_ln<<<NT_, 256, 0, stream>>>(t32, t32, nullptr, nullptr, 1);
  k_rowdot<<<NT_, 256, 0, stream>>>(t32, dlin_w, dlin_b, x, dist, 1);

  // ---- margin normalization ----
  k_zero<<<1, 32, 0, stream>>>(gmax);
  k_margin<<<NT_, L_, 0, stream>>>(height, dist, x, gmax);
  k_dsub<<<(NT_ + 255)/256, 256, 0, stream>>>(dist, gmax);

  // ---- block decomposition + head distribution ----
  k_esum<<<B_, L_, 0, stream>>>(height, scaler, evec, ce);
  k_bmlbmr<<<NT_, L_, 0, stream>>>(height, dist, scaler, bml, bmr);
  k_head<<<NT_, L_, 0, stream>>>(bml, bmr, evec, ce, head);

  // ---- transformer (restart from embeddings) ----
  k_embed<<<NH32, 256, 0, stream>>>(x, emb, hT, maskNeg);
  for (int i = 0; i < NL_; ++i) {
    // attention block
    k_ln<<<NT_, 256, 0, stream>>>(hT, z32, aln_g + i*H_, aln_b + i*H_, 0);
    CVT(z32, buf16, NT_*H_);
    TCVT(qw + (size_t)i*H_*H_, WtQ, H_, H_, 1);
    TCVT(kw + (size_t)i*H_*H_, WtK, H_, H_, 1);
    TCVT(vw + (size_t)i*H_*H_, WtV, H_, H_, 1);
    GEMM(buf16, H_, WtQ, H_, qb + i*H_, q32, H_, NT_, H_, H_, 0);
    GEMM(buf16, H_, WtK, H_, kb + i*H_, k32, H_, NT_, H_, H_, 0);
    GEMM(buf16, H_, WtV, H_, vb + i*H_, v32, H_, NT_, H_, H_, 0);
    k_pack_qkv<<<NH32, 256, 0, stream>>>(q32, k32, v32, qh, kh, vT);
    {
      int waves = B_*NH_*(L_/16)*(L_/64);
      k_attn_score<<<(waves*32 + 255)/256, 256, 0, stream>>>(qh, kh, head, maskNeg,
                                                             relw, i, w16);
    }
    {
      int waves = B_*NH_*(L_/16);
      k_attn_out<<<(waves*32 + 255)/256, 256, 0, stream>>>(w16, vT, o32);
    }
    CVT(o32, buf16, NT_*H_);
    TCVT(ow + (size_t)i*H_*H_, WtO, H_, H_, 1);
    GEMM(buf16, H_, WtO, H_, ob + i*H_, hT, H_, NT_, H_, H_, 1);   // residual add
    // feed-forward block
    k_ln<<<NT_, 256, 0, stream>>>(hT, z32, fln_g + i*H_, fln_b + i*H_, 0);
    CVT(z32, buf16, NT_*H_);
    TCVT(fw1 + (size_t)i*H_*FF_, WtF1, H_, FF_, 1);
    GEMM(buf16, H_, WtF1, H_, fb1 + i*FF_, u32, FF_, NT_, FF_, H_, 2); // leaky_relu
    CVT(u32, buf16, NT_*FF_);
    TCVT(fw2 + (size_t)i*FF_*H_, WtF2, FF_, H_, 1);
    GEMM(buf16, FF_, WtF2, FF_, fb2 + i*H_, hT, H_, NT_, H_, FF_, 1); // residual add
  }

  // ---- final LN + tied-embedding logits ----
  k_ln<<<NT_, 256, 0, stream>>>(hT, z32, fing, finb, 0);
  CVT(z32, buf16, NT_*H_);
  GEMM(buf16, H_, emb16, H_, obias, out, V_, NT_, V_, H_, 0);
}
